// ParallelLSTMModel_87179246174643
// MI455X (gfx1250) — compile-verified, matrix operand order checked by reading, not run on hardware
//
#include <hip/hip_runtime.h>
#include <hip/hip_bf16.h>

#define HH    128
#define TT    50
#define BB    4096
#define NC    21
#define GATES 512           // 4*H
#define CW    12800         // 2*T*H (combined width)

typedef __attribute__((ext_vector_type(16))) __bf16 v16bf;
typedef __attribute__((ext_vector_type(8)))  __bf16 v8bf;
typedef __attribute__((ext_vector_type(8)))  float  v8f;

// Load a 16-element bf16 fragment: 8 bf16 at p0, 8 bf16 at p0+16 (two 16B runs).
// Matches the 16-bit A/B operand layout: per lane, elements 0..7 -> k = hi*8+e,
// elements 8..15 -> k = 16 + hi*8 + (e-8) within a 32-wide K tile.
static __device__ __forceinline__ v16bf ld_frag16(const __bf16* p0) {
  v8bf lo = *(const v8bf*)p0;
  v8bf hh = *(const v8bf*)(p0 + 16);
  v16bf r;
#pragma unroll
  for (int e = 0; e < 8; ++e) { r[e] = lo[e]; r[e + 8] = hh[e]; }
  return r;
}

// Same fragment shape but gathered from f32 memory with on-the-fly bf16 convert.
static __device__ __forceinline__ v16bf ld_frag16_f32(const float* p0) {
  v16bf r;
#pragma unroll
  for (int e = 0; e < 8; ++e) { r[e] = (__bf16)p0[e]; r[e + 8] = (__bf16)p0[16 + e]; }
  return r;
}

static __device__ __forceinline__ float sigm(float x) {
  return 1.0f / (1.0f + __expf(-x));
}

// ---------------------------------------------------------------------------
// Fused 2-layer LSTM scan. One workgroup = 16 batch rows, 16 waves (512 thr).
// Wave (ww = w&7, gp = w>>3): ww selects h columns [16*ww,16*ww+16); gp selects
// the gate pair (0 -> i,f ; 1 -> g,o). Each wave owns 2 complete gate tiles
// (full K), so its persistent layer-0 weight slice is only 8 fragments
// (64 VGPRs) -> no spills. Gates are exchanged through LDS in C-layout for the
// elementwise step; wave gp handles accumulator regs [4gp,4gp+4) (its half of
// the rows) and keeps the matching c-state slice.
// ---------------------------------------------------------------------------
__global__ void __launch_bounds__(512, 1)
lstm_fused_kernel(const float* __restrict__ x_time, const float* __restrict__ x_freq,
                  const float* __restrict__ tWih0, const float* __restrict__ tWhh0,
                  const float* __restrict__ tbih0, const float* __restrict__ tbhh0,
                  const float* __restrict__ tWih1, const float* __restrict__ tWhh1,
                  const float* __restrict__ tbih1, const float* __restrict__ tbhh1,
                  const float* __restrict__ fWih0, const float* __restrict__ fWhh0,
                  const float* __restrict__ fbih0, const float* __restrict__ fbhh0,
                  const float* __restrict__ fWih1, const float* __restrict__ fWhh1,
                  const float* __restrict__ fbih1, const float* __restrict__ fbhh1,
                  float* __restrict__ combined)
{
  __shared__ __align__(16) __bf16 sW1i[GATES * HH];   // 128 KB layer-1 input weights
  __shared__ __align__(16) __bf16 sW1h[GATES * HH];   // 128 KB layer-1 recurrent weights
  __shared__ __align__(16) __bf16 sH0[16 * HH];       // 4 KB  layer-0 h tile (row-major 16x128)
  __shared__ __align__(16) __bf16 sH1[16 * HH];       // 4 KB  layer-1 h tile
  __shared__ __align__(16) float  sX[16 * 2 * TT];    // 6.4 KB raw inputs [row][ch][t]
  __shared__ __align__(16) float  sG[8 * 4 * 8 * 32]; // 32 KB gate tiles in C-layout

  const int br = blockIdx.y;
  const float* Xin  = br ? x_freq : x_time;
  const int    cin  = br ? 2 : 1;
  const float* Wih0 = br ? fWih0 : tWih0;
  const float* Whh0 = br ? fWhh0 : tWhh0;
  const float* bih0 = br ? fbih0 : tbih0;
  const float* bhh0 = br ? fbhh0 : tbhh0;
  const float* Wih1 = br ? fWih1 : tWih1;
  const float* Whh1 = br ? fWhh1 : tWhh1;
  const float* bih1 = br ? fbih1 : tbih1;
  const float* bhh1 = br ? fbhh1 : tbhh1;

  const int tid  = threadIdx.x;
  const int lane = tid & 31;
  const int wave = tid >> 5;
  const int ww   = wave & 7;           // h column slice
  const int gp   = wave >> 3;          // gate pair: 0 -> {i,f}, 1 -> {g,o}
  const int n    = lane & 15;
  const int hi   = lane >> 4;
  const int base16 = blockIdx.x * 16;

  // ---- prologue: fill LDS -------------------------------------------------
  for (int i = tid; i < GATES * HH; i += 512) {
    sW1i[i] = (__bf16)Wih1[i];
    sW1h[i] = (__bf16)Whh1[i];
  }
  for (int i = tid; i < 16 * HH; i += 512) {
    sH0[i] = (__bf16)0.0f;
    sH1[i] = (__bf16)0.0f;
  }
  for (int i = tid; i < 16 * 2 * TT; i += 512) {
    int row = i / (2 * TT);
    int rem = i - row * (2 * TT);
    int c   = rem / TT;
    int t   = rem - c * TT;
    float v = 0.0f;
    if (c < cin) v = Xin[((size_t)(base16 + row) * cin + c) * TT + t];
    sX[i] = v;
  }

  // ---- per-lane constants for this wave's 2 gates -------------------------
  float b0c[2], b1c[2], wx0[2], wx1[2];
#pragma unroll
  for (int j = 0; j < 2; ++j) {
    int g  = gp * 2 + j;
    int gc = g * 128 + ww * 16 + n;
    b0c[j] = bih0[gc] + bhh0[gc];
    b1c[j] = bih1[gc] + bhh1[gc];
    wx0[j] = Wih0[(size_t)gc * cin + 0];
    wx1[j] = (cin > 1) ? Wih0[(size_t)gc * cin + 1] : 0.0f;
  }

  // ---- layer-0 recurrent weight fragments: 8 frags = 64 VGPRs, resident ---
  v16bf wf0[8]; // [j*4 + kt]
#pragma unroll
  for (int j = 0; j < 2; ++j) {
    const int gc = (gp * 2 + j) * 128 + ww * 16 + n;
#pragma unroll
    for (int kt = 0; kt < 4; ++kt) {
      v16bf f;
#pragma unroll
      for (int e = 0; e < 16; ++e) {
        int k = kt * 32 + ((e < 8) ? (hi * 8 + e) : (16 + hi * 8 + (e - 8)));
        f[e] = (__bf16)Whh0[(size_t)gc * HH + k];
      }
      wf0[j * 4 + kt] = f;
    }
  }

  float c0h[4] = {0.f, 0.f, 0.f, 0.f};   // c-state, regs [4gp,4gp+4) of the tile
  float c1h[4] = {0.f, 0.f, 0.f, 0.f};

  __syncthreads();

  // ------------------------------ time scan --------------------------------
#pragma unroll 1
  for (int t = 0; t < TT; ++t) {
    // ---- layer 0: gates = bias + h0_prev @ Whh0^T + x-projection ----------
    v8f a0g[2];
#pragma unroll
    for (int j = 0; j < 2; ++j)
#pragma unroll
      for (int r = 0; r < 8; ++r) a0g[j][r] = b0c[j];
#pragma unroll
    for (int kt = 0; kt < 4; ++kt) {
      v16bf af = ld_frag16(&sH0[n * HH + kt * 32 + hi * 8]);
#pragma unroll
      for (int j = 0; j < 2; ++j)
        a0g[j] = __builtin_amdgcn_wmma_f32_16x16x32_bf16(
            false, af, false, wf0[j * 4 + kt], (short)0, a0g[j], false, false);
    }
    {
      float xv0[8], xv1[8];
#pragma unroll
      for (int r = 0; r < 8; ++r) {
        int ml = r + 8 * hi;
        xv0[r] = sX[ml * (2 * TT) + t];
        xv1[r] = sX[ml * (2 * TT) + TT + t];
      }
#pragma unroll
      for (int j = 0; j < 2; ++j)
#pragma unroll
        for (int r = 0; r < 8; ++r)
          a0g[j][r] += wx0[j] * xv0[r] + wx1[j] * xv1[r];
    }
    // publish gate tiles (C-layout) for the elementwise exchange
#pragma unroll
    for (int j = 0; j < 2; ++j)
#pragma unroll
      for (int r = 0; r < 8; ++r)
        sG[((ww * 4 + gp * 2 + j) * 8 + r) * 32 + lane] = a0g[j][r];
    __syncthreads();   // B1: sG complete

    // elementwise on this wave's half of the rows
#pragma unroll
    for (int rr = 0; rr < 4; ++rr) {
      int r = gp * 4 + rr;
      float ig = sigm (sG[((ww * 4 + 0) * 8 + r) * 32 + lane]);
      float fg = sigm (sG[((ww * 4 + 1) * 8 + r) * 32 + lane]);
      float gg = tanhf(sG[((ww * 4 + 2) * 8 + r) * 32 + lane]);
      float og = sigm (sG[((ww * 4 + 3) * 8 + r) * 32 + lane]);
      float c  = fg * c0h[rr] + ig * gg;
      c0h[rr]  = c;
      float h  = og * tanhf(c);
      sH0[(r + 8 * hi) * HH + ww * 16 + n] = (__bf16)h;
    }
    __syncthreads();   // B2: sH0 now holds h0(t); sG reads done

    // ---- layer 1: input pass over sH0, recurrent pass over sH1 ------------
    v8f a1g[2];
#pragma unroll
    for (int j = 0; j < 2; ++j)
#pragma unroll
      for (int r = 0; r < 8; ++r) a1g[j][r] = b1c[j];
#pragma unroll
    for (int kt = 0; kt < 4; ++kt) {
      v16bf af = ld_frag16(&sH0[n * HH + kt * 32 + hi * 8]);
#pragma unroll
      for (int j = 0; j < 2; ++j) {
        v16bf bI = ld_frag16(&sW1i[((gp * 2 + j) * 128 + ww * 16 + n) * HH + kt * 32 + hi * 8]);
        a1g[j] = __builtin_amdgcn_wmma_f32_16x16x32_bf16(
            false, af, false, bI, (short)0, a1g[j], false, false);
      }
    }
#pragma unroll
    for (int kt = 0; kt < 4; ++kt) {
      v16bf af = ld_frag16(&sH1[n * HH + kt * 32 + hi * 8]);
#pragma unroll
      for (int j = 0; j < 2; ++j) {
        v16bf bR = ld_frag16(&sW1h[((gp * 2 + j) * 128 + ww * 16 + n) * HH + kt * 32 + hi * 8]);
        a1g[j] = __builtin_amdgcn_wmma_f32_16x16x32_bf16(
            false, af, false, bR, (short)0, a1g[j], false, false);
      }
    }
#pragma unroll
    for (int j = 0; j < 2; ++j)
#pragma unroll
      for (int r = 0; r < 8; ++r)
        sG[((ww * 4 + gp * 2 + j) * 8 + r) * 32 + lane] = a1g[j][r];
    __syncthreads();   // B3: sG complete; sH1 reads done

#pragma unroll
    for (int rr = 0; rr < 4; ++rr) {
      int r = gp * 4 + rr;
      float ig = sigm (sG[((ww * 4 + 0) * 8 + r) * 32 + lane]);
      float fg = sigm (sG[((ww * 4 + 1) * 8 + r) * 32 + lane]);
      float gg = tanhf(sG[((ww * 4 + 2) * 8 + r) * 32 + lane]);
      float og = sigm (sG[((ww * 4 + 3) * 8 + r) * 32 + lane]);
      float c  = fg * c1h[rr] + ig * gg;
      c1h[rr]  = c;
      float h  = og * tanhf(c);
      int  m   = r + 8 * hi;
      sH1[m * HH + ww * 16 + n] = (__bf16)h;
      combined[(size_t)(base16 + m) * CW +
               (size_t)br * (TT * HH) + (size_t)t * HH + ww * 16 + n] = h;
    }
    __syncthreads();   // B4: sH1 published; sG reads done before next overwrite
  }
}

// ---------------------------------------------------------------------------
// FC head: out[b, cls] = combined[b, :] . fcW[cls, :] + fcb[cls]
// One workgroup = 16 rows; 8 waves split K (400 k-tiles -> 50 each);
// N = 21 padded to two 16-wide tiles; flat LDS sum across the 8 waves.
// ---------------------------------------------------------------------------
__global__ void __launch_bounds__(256, 1)
fc_head_kernel(const float* __restrict__ combined, const float* __restrict__ fcW,
               const float* __restrict__ fcb, float* __restrict__ out)
{
  __shared__ __align__(16) float sRed[8 * 2 * 8 * 32];  // 16 KB partials

  const int tid  = threadIdx.x;
  const int lane = tid & 31;
  const int wave = tid >> 5;
  const int n    = lane & 15;
  const int hi   = lane >> 4;
  const int base = blockIdx.x * 16;

  v8f acc[2] = {{}, {}};

  for (int j = 0; j < 50; ++j) {
    int kt = wave * 50 + j;
    int kc = kt * 32 + hi * 8;
    v16bf af = ld_frag16_f32(combined + (size_t)(base + n) * CW + kc);
#pragma unroll
    for (int nt = 0; nt < 2; ++nt) {
      int nr = nt * 16 + n;
      v16bf bf;
      if (nr < NC) {
        bf = ld_frag16_f32(fcW + (size_t)nr * CW + kc);
      } else {
#pragma unroll
        for (int e = 0; e < 16; ++e) bf[e] = (__bf16)0.0f;
      }
      acc[nt] = __builtin_amdgcn_wmma_f32_16x16x32_bf16(
          false, af, false, bf, (short)0, acc[nt], false, false);
    }
  }

#pragma unroll
  for (int nt = 0; nt < 2; ++nt)
#pragma unroll
    for (int r = 0; r < 8; ++r)
      sRed[((wave * 2 + nt) * 8 + r) * 32 + lane] = acc[nt][r];
  __syncthreads();

  for (int j = tid; j < 512; j += 256) {
    int nt  = j >> 8;
    int rem = j & 255;
    int r   = rem >> 5;
    int l2  = rem & 31;
    float s = 0.0f;
#pragma unroll
    for (int w = 0; w < 8; ++w) s += sRed[((w * 2 + nt) * 8 + r) * 32 + l2];
    int mm = r + 8 * (l2 >> 4);
    int nn = nt * 16 + (l2 & 15);
    if (nn < NC) out[(size_t)(base + mm) * NC + nn] = s + fcb[nn];
  }
}

// ---------------------------------------------------------------------------
extern "C" void kernel_launch(void* const* d_in, const int* in_sizes, int n_in,
                              void* d_out, int out_size, void* d_ws, size_t ws_size,
                              hipStream_t stream) {
  const float* x_time = (const float*)d_in[0];
  const float* x_freq = (const float*)d_in[1];
  const float* tWih0  = (const float*)d_in[2];
  const float* tWhh0  = (const float*)d_in[3];
  const float* tbih0  = (const float*)d_in[4];
  const float* tbhh0  = (const float*)d_in[5];
  const float* tWih1  = (const float*)d_in[6];
  const float* tWhh1  = (const float*)d_in[7];
  const float* tbih1  = (const float*)d_in[8];
  const float* tbhh1  = (const float*)d_in[9];
  const float* fWih0  = (const float*)d_in[10];
  const float* fWhh0  = (const float*)d_in[11];
  const float* fbih0  = (const float*)d_in[12];
  const float* fbhh0  = (const float*)d_in[13];
  const float* fWih1  = (const float*)d_in[14];
  const float* fWhh1  = (const float*)d_in[15];
  const float* fbih1  = (const float*)d_in[16];
  const float* fbhh1  = (const float*)d_in[17];
  const float* fcW    = (const float*)d_in[18];
  const float* fcb    = (const float*)d_in[19];

  float* out      = (float*)d_out;
  float* combined = out + (size_t)BB * NC;

  dim3 g1(BB / 16, 2), b1(512);
  lstm_fused_kernel<<<g1, b1, 0, stream>>>(
      x_time, x_freq,
      tWih0, tWhh0, tbih0, tbhh0, tWih1, tWhh1, tbih1, tbhh1,
      fWih0, fWhh0, fbih0, fbhh0, fWih1, fWhh1, fbih1, fbhh1,
      combined);

  fc_head_kernel<<<dim3(BB / 16), dim3(256), 0, stream>>>(combined, fcW, fcb, out);
}
